// MLA_1005022347534
// MI455X (gfx1250) — matrix operand order repaired and checked by reading, not compiled, not run on hardware
//
#include <hip/hip_runtime.h>

typedef __attribute__((ext_vector_type(16))) _Float16 v16h;
typedef __attribute__((ext_vector_type(8)))  _Float16 v8h;
typedef __attribute__((ext_vector_type(8)))  float    v8f;

#define TSEQ 2048
#define NH   16
#define DKQ  128     // DK
#define DR   64      // DHR
#define DQK  192     // DK + DHR
#define CDIM 2048    // MODEL_DIM
#define LAT  512

static __device__ __forceinline__ v8f zero8() {
  v8f z;
#pragma unroll
  for (int i = 0; i < 8; ++i) z[i] = 0.0f;
  return z;
}

// Async DMA: global -> LDS, 16 bytes per lane, tracked by ASYNCcnt (ISA 08 §4).
// LDS byte offset = low 32 bits of the generic pointer (flat->LDS aperture rule).
static __device__ __forceinline__ void async_ld16(const _Float16* g, _Float16* l) {
  asm volatile("global_load_async_to_lds_b128 %0, %1, off"
               :
               : "v"((unsigned)(uintptr_t)l), "v"((unsigned long long)(uintptr_t)g)
               : "memory");
}
static __device__ __forceinline__ void wait_async0() {
  asm volatile("s_wait_asynccnt 0" ::: "memory");
}

// A fragment 16x32 f16: lane m=L&15, g=L>>4;
// elems 0-15 -> k = g*16+e (contiguous, matches B)
static __device__ __forceinline__ v16h load_a_frag(const _Float16* A, int lda) {
  int lane = threadIdx.x & 31;
  const _Float16* p = A + (size_t)(lane & 15) * lda + ((lane >> 4) << 4);
  v8h lo = *(const v8h*)p;
  v8h hi = *(const v8h*)(p + 8);
  v16h r;
#pragma unroll
  for (int i = 0; i < 8; ++i) { r[i] = lo[i]; r[i + 8] = hi[i]; }
  return r;
}

// B fragment 32x16 f16: lane holds column n=L&15, K run = (L>>4)*16 + e.
static __device__ __forceinline__ v16h load_b_frag(const _Float16* Bt, int ldb) {
  int lane = threadIdx.x & 31;
  return *(const v16h*)(Bt + (size_t)(lane & 15) * ldb + ((lane >> 4) << 4));
}

static __device__ __forceinline__ v8f wmma16(v16h a, v16h b, v8f c) {
  return __builtin_amdgcn_wmma_f32_16x16x32_f16(false, a, false, b, (short)0, c,
                                                false, false);
}

// ---------- single-wave 16x64 tile (used only for tiny kR GEMM, N=64) ----------
static __device__ __forceinline__ void gemm_tile(const _Float16* A, int lda,
                                                 const _Float16* Bt, int ldb,
                                                 int K, v8f acc[4]) {
  for (int k = 0; k < K; k += 32) {
    v16h a = load_a_frag(A + k, lda);
#pragma unroll
    for (int f = 0; f < 4; ++f) {
      v16h b = load_b_frag(Bt + (size_t)(f * 16) * ldb + k, ldb);
      acc[f] = wmma16(a, b, acc[f]);
    }
  }
}

// ---------- block GEMM: 256 thr (8 waves), 128x128 block tile, 32x64 per wave ----
// LDS double-buffered k-slices of A (128x32) and Bt (128x32). Next tile is fetched
// with async global->LDS DMA issued BEFORE this tile's WMMAs; s_wait_asynccnt 0
// lands after the compute, so the DMA fully overlaps matrix math.
static __device__ __forceinline__ void block_gemm(
    const _Float16* __restrict__ A, int lda,          // pre-offset to row0
    const _Float16* __restrict__ Bt, int ldb, int K,  // pre-offset to col0
    _Float16* As, _Float16* Bs,                       // each 2*4096 halfs
    v8f acc[8]) {                                     // acc[f]=rows0-15, acc[4+f]=rows16-31
  int tid = threadIdx.x;
  int wid = tid >> 5, wr = wid >> 1, wc = wid & 1;
  int r0 = tid >> 2, o0 = (tid & 3) << 3;             // chunk 0
  int r1 = r0 + 64;                                   // chunk 1

  auto issue = [&](int buf, int k) {
    async_ld16(A + (size_t)r0 * lda + k + o0, As + buf * 4096 + r0 * 32 + o0);
    async_ld16(A + (size_t)r1 * lda + k + o0, As + buf * 4096 + r1 * 32 + o0);
    async_ld16(Bt + (size_t)r0 * ldb + k + o0, Bs + buf * 4096 + r0 * 32 + o0);
    async_ld16(Bt + (size_t)r1 * ldb + k + o0, Bs + buf * 4096 + r1 * 32 + o0);
  };

  issue(0, 0);
  wait_async0();
  __syncthreads();
  int nsteps = K >> 5;
  for (int i = 0; i < nsteps; ++i) {
    int buf = i & 1;
    if (i + 1 < nsteps) issue(buf ^ 1, (i + 1) << 5);
    const _Float16* Ab = As + buf * 4096 + (wr * 32) * 32;
    const _Float16* Bb = Bs + buf * 4096 + (wc * 64) * 32;
    v16h a0 = load_a_frag(Ab, 32);
    v16h a1 = load_a_frag(Ab + 16 * 32, 32);
#pragma unroll
    for (int f = 0; f < 4; ++f) {
      v16h bb = load_b_frag(Bb + f * 16 * 32, 32);
      acc[f]     = wmma16(a0, bb, acc[f]);
      acc[4 + f] = wmma16(a1, bb, acc[4 + f]);
    }
    wait_async0();
    __syncthreads();
  }
}

// ---------------- conversion kernels ----------------

__global__ __launch_bounds__(256) void cvt_f16_kernel(const float* __restrict__ in,
                                                      _Float16* __restrict__ out,
                                                      int n) {
  int i = (blockIdx.x * 256 + threadIdx.x) * 4;
  if (i + 3 < n) {
    float4 v = *(const float4*)(in + i);
    out[i + 0] = (_Float16)v.x;
    out[i + 1] = (_Float16)v.y;
    out[i + 2] = (_Float16)v.z;
    out[i + 3] = (_Float16)v.w;
  }
}

// W (K x N, row-major f32) -> Wt (N x K, row-major f16)
__global__ __launch_bounds__(256) void transpose_w_kernel(const float* __restrict__ W,
                                                          _Float16* __restrict__ Wt,
                                                          int K, int N) {
  int idx = blockIdx.x * 256 + threadIdx.x;
  if (idx >= N * K) return;
  int n = idx / K, k = idx - n * K;
  Wt[idx] = (_Float16)W[(size_t)k * N + n];
}

// ---------------- GEMM kernels (block-tiled) ----------------
// epilogue index: frag fi (0..7):
//   row = row0 + wr*32 + (fi>=4 ? 16:0) + g*8 + j ; col = col0 + wc*64 + (fi&3)*16 + n

// down = x @ W_down + b ; f16 (8192 x 1024): cols 0..511 = cKV, 512..1023 = cq
__global__ __launch_bounds__(256) void gemm_down_kernel(const _Float16* __restrict__ xh,
                                                        const _Float16* __restrict__ Wt,
                                                        const float* __restrict__ bias,
                                                        _Float16* __restrict__ out) {
  __shared__ _Float16 As[2 * 4096];
  __shared__ _Float16 Bs[2 * 4096];
  int col0 = blockIdx.x * 128, row0 = blockIdx.y * 128;
  v8f acc[8];
#pragma unroll
  for (int i = 0; i < 8; ++i) acc[i] = zero8();
  block_gemm(xh + (size_t)row0 * CDIM, CDIM, Wt + (size_t)col0 * CDIM, CDIM, CDIM,
             As, Bs, acc);
  int lane = threadIdx.x & 31, n = lane & 15, g = lane >> 4;
  int wid = threadIdx.x >> 5, wr = wid >> 1, wc = wid & 1;
#pragma unroll
  for (int fi = 0; fi < 8; ++fi)
#pragma unroll
    for (int j = 0; j < 8; ++j) {
      int row = row0 + wr * 32 + ((fi >> 2) << 4) + g * 8 + j;
      int col = col0 + wc * 64 + (fi & 3) * 16 + n;
      out[(size_t)row * 1024 + col] = (_Float16)(acc[fi][j] + bias[col]);
    }
}

// qR GEMM (N=1024) with fused RoPE, scaled, scattered into qbuf dims 128..191
__global__ __launch_bounds__(256) void gemm_qr_kernel(const _Float16* __restrict__ xh,
                                                      const _Float16* __restrict__ Wt,
                                                      const float* __restrict__ bias,
                                                      _Float16* __restrict__ qbuf) {
  __shared__ _Float16 As[2 * 4096];
  __shared__ _Float16 Bs[2 * 4096];
  int col0 = blockIdx.x * 128, row0 = blockIdx.y * 128;
  v8f acc[8];
#pragma unroll
  for (int i = 0; i < 8; ++i) acc[i] = zero8();
  block_gemm(xh + (size_t)row0 * CDIM, CDIM, Wt + (size_t)col0 * CDIM, CDIM, CDIM,
             As, Bs, acc);
  int lane = threadIdx.x & 31, n = lane & 15, g = lane >> 4;
  int wid = threadIdx.x >> 5, wr = wid >> 1, wc = wid & 1;
  const float qs = 0.08838834764831845f;  // 1/sqrt(128)
#pragma unroll
  for (int fi = 0; fi < 8; ++fi)
#pragma unroll
    for (int j = 0; j < 8; ++j) {
      int row = row0 + wr * 32 + ((fi >> 2) << 4) + g * 8 + j;
      int col = col0 + wc * 64 + (fi & 3) * 16 + n;
      float val = acc[fi][j] + bias[col];
      float partner = __shfl_xor(val, 1);  // col^1 lives in lane n^1
      int d = col & 63, h = col >> 6;
      int tt = row & (TSEQ - 1), bb = row >> 11;
      float theta = expf(-(float)(d >> 1) * (9.210340371976184f / 32.0f));
      float ang = (float)tt * theta;
      float rot = (d & 1) ? partner : -partner;
      float res = val * cosf(ang) + rot * sinf(ang);
      qbuf[((size_t)((bb * NH + h) * TSEQ + tt)) * DQK + DKQ + d] =
          (_Float16)(res * qs);
    }
}

// v_k = cKV @ W_uKV + b (N=4096); v -> vT (b,h,d,t), k -> kbuf (b,h,t,192) dims 0..127
__global__ __launch_bounds__(256) void gemm_ukv_kernel(const _Float16* __restrict__ cKV,
                                                       const _Float16* __restrict__ Wt,
                                                       const float* __restrict__ bias,
                                                       _Float16* __restrict__ vT,
                                                       _Float16* __restrict__ kbuf) {
  __shared__ _Float16 As[2 * 4096];
  __shared__ _Float16 Bs[2 * 4096];
  int col0 = blockIdx.x * 128, row0 = blockIdx.y * 128;
  v8f acc[8];
#pragma unroll
  for (int i = 0; i < 8; ++i) acc[i] = zero8();
  block_gemm(cKV + (size_t)row0 * 1024, 1024, Wt + (size_t)col0 * LAT, LAT, LAT,
             As, Bs, acc);
  int lane = threadIdx.x & 31, n = lane & 15, g = lane >> 4;
  int wid = threadIdx.x >> 5, wr = wid >> 1, wc = wid & 1;
#pragma unroll
  for (int fi = 0; fi < 8; ++fi)
#pragma unroll
    for (int j = 0; j < 8; ++j) {
      int row = row0 + wr * 32 + ((fi >> 2) << 4) + g * 8 + j;
      int col = col0 + wc * 64 + (fi & 3) * 16 + n;
      float val = acc[fi][j] + bias[col];
      int h = col >> 8, jj = col & 255;
      int tt = row & (TSEQ - 1), bb = row >> 11;
      if (jj < DKQ)
        vT[((size_t)((bb * NH + h) * DKQ + jj)) * TSEQ + tt] = (_Float16)val;
      else
        kbuf[((size_t)((bb * NH + h) * TSEQ + tt)) * DQK + (jj - DKQ)] = (_Float16)val;
    }
}

// q = cq @ W_uQ + b (N=2048, scaled) -> qbuf dims 0..127
__global__ __launch_bounds__(256) void gemm_uq_kernel(const _Float16* __restrict__ cq,
                                                      const _Float16* __restrict__ Wt,
                                                      const float* __restrict__ bias,
                                                      _Float16* __restrict__ qbuf) {
  __shared__ _Float16 As[2 * 4096];
  __shared__ _Float16 Bs[2 * 4096];
  int col0 = blockIdx.x * 128, row0 = blockIdx.y * 128;
  v8f acc[8];
#pragma unroll
  for (int i = 0; i < 8; ++i) acc[i] = zero8();
  block_gemm(cq + (size_t)row0 * 1024, 1024, Wt + (size_t)col0 * LAT, LAT, LAT,
             As, Bs, acc);
  int lane = threadIdx.x & 31, n = lane & 15, g = lane >> 4;
  int wid = threadIdx.x >> 5, wr = wid >> 1, wc = wid & 1;
  const float qs = 0.08838834764831845f;
#pragma unroll
  for (int fi = 0; fi < 8; ++fi)
#pragma unroll
    for (int j = 0; j < 8; ++j) {
      int row = row0 + wr * 32 + ((fi >> 2) << 4) + g * 8 + j;
      int col = col0 + wc * 64 + (fi & 3) * 16 + n;
      float val = (acc[fi][j] + bias[col]) * qs;
      int h = col >> 7, jj = col & 127;
      int tt = row & (TSEQ - 1), bb = row >> 11;
      qbuf[((size_t)((bb * NH + h) * TSEQ + tt)) * DQK + jj] = (_Float16)val;
    }
}

// final: out = attn @ W_out + b_out (f32 store, N=2048)
__global__ __launch_bounds__(256) void gemm_out_kernel(const _Float16* __restrict__ attn,
                                                       const _Float16* __restrict__ Wt,
                                                       const float* __restrict__ bias,
                                                       float* __restrict__ out) {
  __shared__ _Float16 As[2 * 4096];
  __shared__ _Float16 Bs[2 * 4096];
  int col0 = blockIdx.x * 128, row0 = blockIdx.y * 128;
  v8f acc[8];
#pragma unroll
  for (int i = 0; i < 8; ++i) acc[i] = zero8();
  block_gemm(attn + (size_t)row0 * CDIM, CDIM, Wt + (size_t)col0 * CDIM, CDIM, CDIM,
             As, Bs, acc);
  int lane = threadIdx.x & 31, n = lane & 15, g = lane >> 4;
  int wid = threadIdx.x >> 5, wr = wid >> 1, wc = wid & 1;
#pragma unroll
  for (int fi = 0; fi < 8; ++fi)
#pragma unroll
    for (int j = 0; j < 8; ++j) {
      int row = row0 + wr * 32 + ((fi >> 2) << 4) + g * 8 + j;
      int col = col0 + wc * 64 + (fi & 3) * 16 + n;
      out[(size_t)row * CDIM + col] = acc[fi][j] + bias[col];
    }
}

// kR GEMM (N=64) with fused RoPE, broadcast to all heads: single-wave path
__global__ __launch_bounds__(32) void gemm_kr_kernel(const _Float16* __restrict__ xh,
                                                     const _Float16* __restrict__ Wt,
                                                     const float* __restrict__ bias,
                                                     _Float16* __restrict__ kbuf) {
  int row0 = blockIdx.y * 16;
  v8f acc[4] = {zero8(), zero8(), zero8(), zero8()};
  gemm_tile(xh + (size_t)row0 * CDIM, CDIM, Wt, CDIM, CDIM, acc);
  int lane = threadIdx.x & 31, n = lane & 15, g = lane >> 4;
#pragma unroll
  for (int f = 0; f < 4; ++f)
#pragma unroll
    for (int j = 0; j < 8; ++j) {
      int d = f * 16 + n;
      int row = row0 + g * 8 + j;
      float val = acc[f][j] + bias[d];
      float partner = __shfl_xor(val, 1);
      int tt = row & (TSEQ - 1), bb = row >> 11;
      float theta = expf(-(float)(d >> 1) * (9.210340371976184f / 32.0f));
      float ang = (float)tt * theta;
      float rot = (d & 1) ? partner : -partner;
      _Float16 res = (_Float16)(val * cosf(ang) + rot * sinf(ang));
#pragma unroll
      for (int h = 0; h < NH; ++h)
        kbuf[((size_t)((bb * NH + h) * TSEQ + tt)) * DQK + DKQ + d] = res;
    }
}

// ---------------- flash attention ----------------
// grid (T/128, NH, B), 256 thr (8 waves); block = 128 q rows, wave = 16 q rows.
// K (32x192) and V (128x32) tiles double-buffered in LDS via async global->LDS DMA.
__global__ __launch_bounds__(256) void attn_kernel(const _Float16* __restrict__ qbuf,
                                                   const _Float16* __restrict__ kbuf,
                                                   const _Float16* __restrict__ vT,
                                                   _Float16* __restrict__ attn) {
  __shared__ _Float16 Ks[2 * 6144];   // [buf][key(32)][192]
  __shared__ _Float16 Vs[2 * 4096];   // [buf][d(128)][32]
  __shared__ _Float16 Pl[8 * 512];    // per-wave P scratch [16][32]
  int tid = threadIdx.x;
  int wid = tid >> 5, lane = tid & 31, n = lane & 15, g = lane >> 4;
  int qb = blockIdx.x, h = blockIdx.y, b = blockIdx.z;
  size_t bh = (size_t)(b * NH + h);

  const _Float16* qbase = qbuf + (bh * TSEQ + qb * 128 + wid * 16) * DQK;
  const _Float16* kbase = kbuf + bh * TSEQ * DQK;
  const _Float16* vbase = vT + bh * DKQ * TSEQ;
  _Float16* Pw = Pl + wid * 512;

  auto issue = [&](int buf, int s) {
#pragma unroll
    for (int c = 0; c < 3; ++c)
      async_ld16(kbase + (size_t)s * DQK + (tid + c * 256) * 8,
                 Ks + buf * 6144 + (tid + c * 256) * 8);
#pragma unroll
    for (int c = 0; c < 2; ++c) {
      int p = tid + c * 256, d = p >> 2, o = (p & 3) << 3;
      async_ld16(vbase + (size_t)d * TSEQ + s + o, Vs + buf * 4096 + d * 32 + o);
    }
  };

  v16h qa[6];
#pragma unroll
  for (int c = 0; c < 6; ++c) qa[c] = load_a_frag(qbase + c * 32, DQK);

  v8f O[8];
#pragma unroll
  for (int f = 0; f < 8; ++f) O[f] = zero8();
  float m[8], l[8];
#pragma unroll
  for (int j = 0; j < 8; ++j) { m[j] = -1e30f; l[j] = 0.0f; }

  issue(0, 0);
  wait_async0();
  __syncthreads();

  for (int i = 0; i < TSEQ / 32; ++i) {
    int buf = i & 1;
    if (i + 1 < TSEQ / 32) issue(buf ^ 1, (i + 1) * 32);

    const _Float16* Kb = Ks + buf * 6144;
    const _Float16* Vb = Vs + buf * 4096;

    v8f S0 = zero8(), S1 = zero8();
#pragma unroll
    for (int c = 0; c < 6; ++c) {
      v16h b0 = load_b_frag(Kb + c * 32, DQK);
      v16h b1 = load_b_frag(Kb + 16 * DQK + c * 32, DQK);
      S0 = wmma16(qa[c], b0, S0);
      S1 = wmma16(qa[c], b1, S1);
    }
    float alpha[8];
#pragma unroll
    for (int j = 0; j < 8; ++j) {
      float tm = fmaxf(S0[j], S1[j]);
#pragma unroll
      for (int off = 1; off < 16; off <<= 1) tm = fmaxf(tm, __shfl_xor(tm, off));
      float mn = fmaxf(m[j], tm);
      alpha[j] = __expf(m[j] - mn);
      m[j] = mn;
      float p0 = __expf(S0[j] - mn);
      float p1 = __expf(S1[j] - mn);
      Pw[(g * 8 + j) * 32 + n] = (_Float16)p0;
      Pw[(g * 8 + j) * 32 + 16 + n] = (_Float16)p1;
      float rs = p0 + p1;
#pragma unroll
      for (int off = 1; off < 16; off <<= 1) rs += __shfl_xor(rs, off);
      l[j] = l[j] * alpha[j] + rs;
    }
    asm volatile("s_wait_dscnt 0" ::: "memory");  // P stores before A-frag reads
#pragma unroll
    for (int f = 0; f < 8; ++f)
#pragma unroll
      for (int j = 0; j < 8; ++j) O[f][j] *= alpha[j];
    v16h pa = load_a_frag(Pw, 32);
#pragma unroll
    for (int f = 0; f < 8; ++f) {
      v16h vb = load_b_frag(Vb + f * 16 * 32, 32);
      O[f] = wmma16(pa, vb, O[f]);
    }
    wait_async0();
    __syncthreads();
  }

  float inv[8];
#pragma unroll
  for (int j = 0; j < 8; ++j) inv[j] = 1.0f / l[j];
#pragma unroll
  for (int f = 0; f < 8; ++f)
#pragma unroll
    for (int j = 0; j < 8; ++j) {
      int row = qb * 128 + wid * 16 + g * 8 + j;
      attn[((size_t)(b * TSEQ + row)) * CDIM + h * DKQ + f * 16 + n] =
          (_Float16)(O[f][j] * inv[j]);
    }
}

// ---------------- host launcher ----------------

extern "C" void kernel_launch(void* const* d_in, const int* in_sizes, int n_in,
                              void* d_out, int out_size, void* d_ws, size_t ws_size,
                              hipStream_t stream) {
  const float* x      = (const float*)d_in[0];
  const float* W_down = (const float*)d_in[1];
  const float* b_down = (const float*)d_in[2];
  const float* W_uKV  = (const float*)d_in[3];
  const float* b_uKV  = (const float*)d_in[4];
  const float* W_uQ   = (const float*)d_in[5];
  const float* b_uQ   = (const float*)d_in[6];
  const float* W_kr   = (const float*)d_in[7];
  const float* b_kr   = (const float*)d_in[8];
  const float* W_qr   = (const float*)d_in[9];
  const float* b_qr   = (const float*)d_in[10];
  const float* W_out  = (const float*)d_in[11];
  const float* b_out  = (const float*)d_in[12];
  float* out = (float*)d_out;

  char* w = (char*)d_ws;
  auto alloc = [&](size_t elems) {
    char* p = w;
    w += (elems * 2 + 255) & ~(size_t)255;
    return (_Float16*)p;
  };
  _Float16* xh     = alloc(16777216);  // 8192 x 2048
  _Float16* WtDown = alloc(2097152);   // 1024 x 2048
  _Float16* WtKr   = alloc(131072);    //   64 x 2048
  _Float16* WtQr   = alloc(2097152);   // 1024 x 2048
  _Float16* WtUKV  = alloc(2097152);   // 4096 x 512
  _Float16* WtUQ   = alloc(1048576);   // 2048 x 512
  _Float16* WtOut  = alloc(4194304);   // 2048 x 2048
  _Float16* cKVq   = alloc(8388608);   // 8192 x 1024 (cKV | cq)
  _Float16* qbuf   = alloc(25165824);  // (B,H,T,192)
  _Float16* kbuf   = alloc(25165824);  // (B,H,T,192)
  _Float16* vTb    = alloc(16777216);  // (B,H,128,T)
  _Float16* attnb  = alloc(16777216);  // 8192 x 2048

  cvt_f16_kernel<<<16384, 256, 0, stream>>>(x, xh, 16777216);
  transpose_w_kernel<<<8192, 256, 0, stream>>>(W_down, WtDown, 2048, 1024);
  transpose_w_kernel<<<512, 256, 0, stream>>>(W_kr, WtKr, 2048, 64);
  transpose_w_kernel<<<8192, 256, 0, stream>>>(W_qr, WtQr, 2048, 1024);
  transpose_w_kernel<<<8192, 256, 0, stream>>>(W_uKV, WtUKV, 512, 4096);
  transpose_w_kernel<<<4096, 256, 0, stream>>>(W_uQ, WtUQ, 512, 2048);
  transpose_w_kernel<<<16384, 256, 0, stream>>>(W_out, WtOut, 2048, 2048);

  gemm_down_kernel<<<dim3(8, 64), 256, 0, stream>>>(xh, WtDown, b_down, cKVq);
  gemm_kr_kernel<<<dim3(1, 512), 32, 0, stream>>>(xh, WtKr, b_kr, kbuf);
  gemm_qr_kernel<<<dim3(8, 64), 256, 0, stream>>>(xh, WtQr, b_qr, qbuf);
  gemm_ukv_kernel<<<dim3(32, 64), 256, 0, stream>>>(cKVq, WtUKV, b_uKV, vTb, kbuf);
  gemm_uq_kernel<<<dim3(16, 64), 256, 0, stream>>>(cKVq + 512, WtUQ, b_uQ, qbuf);

  attn_kernel<<<dim3(16, 16, 4), 256, 0, stream>>>(qbuf, kbuf, vTb, attnb);

  gemm_out_kernel<<<dim3(16, 64), 256, 0, stream>>>(attnb, WtOut, b_out, out);
}
